// Model_54262616818215
// MI455X (gfx1250) — matrix operand order repaired and checked by reading, not compiled, not run on hardware
//
#include <hip/hip_runtime.h>
#include <hip/hip_bf16.h>
#include <math.h>

// ---------------------------------------------------------------------------
// GAT model fused for MI455X (gfx1250, wave32, WMMA bf16 16x16x32).
// Only t=0 of the concatenated features feeds the LSTM head, so we compute
// exactly the 6 GAT1 heads / 1 GAT2 head that matter.  BatchNorm is folded
// into the GEMM weights (scale into B, shift into a bias vector).
// ---------------------------------------------------------------------------

typedef float  v8f   __attribute__((ext_vector_type(8)));
typedef __bf16 v16bf __attribute__((ext_vector_type(16)));

#define NSAMP 32768
#define NV 7
#define NC 6
#define NT 30
#define SAMP_FLOATS 1260        /* NC*NT*NV */
#define BN_CNT (NSAMP * NT)     /* stats count per channel */

/* ws layout (float offsets) */
#define OFF_SCALE 0             /* 42 */
#define OFF_SHIFT 42            /* 42 */
#define OFF_BIAS  84            /* 7*128 = 896 */
#define OFF_ADJ1  980           /* 6*49  = 294 */
#define OFF_ADJ2  1274          /* 49 */
#define OFF_A1    1323          /* 6*18  = 108 */
#define OFF_A2    1431          /* 20 */
#define OFF_GB    1451          /* 96 */
/* byte offsets for bf16 / partial regions (32-byte aligned) */
#define OFFB_WSB   6208                      /* 7*6*8*512 bf16 = 344064 B */
#define OFFB_WIH   (OFFB_WSB + 344064)       /* 18*512 bf16   = 18432 B  */
#define OFFB_PART  (OFFB_WIH + 18432)        /* 256*84 f32    = 86016 B  */
/* total ws usage ~455 KB */

__device__ __forceinline__ int kmapA(int i, int L) {           // 16-bit A 16x32
    return i + ((i & 8) ? 8 : 0) + ((L & 16) ? 8 : 0);
}
__device__ __forceinline__ int kmapB(int i, int L) {           // 16-bit B 32x16
    return i + ((i & 8) ? 8 : 0) + ((L & 16) ? 8 : 0);
}
__device__ __forceinline__ float leakyf(float x) { return x >= 0.f ? x : 0.2f * x; }
__device__ __forceinline__ float eluf(float x)   { return x >= 0.f ? x : expm1f(x); }
__device__ __forceinline__ float sigmf(float x)  { return 1.f / (1.f + expf(-x)); }
__device__ __forceinline__ int headOf(int g)     { return (g < 3) ? 3 * g : 3 * g + 1; }

// ---------------------------------------------------------------------------
// Kernel 1: BN statistics.  256 blocks x 128 samples, deterministic partials.
// ---------------------------------------------------------------------------
__global__ void k_stats(const float* __restrict__ x, float* __restrict__ part)
{
    extern __shared__ float lds[];          // 16*1260 floats
    const int tid = threadIdx.x, blk = blockIdx.x;
    float sum = 0.f, sq = 0.f;
    const int ch = tid / 3, p = tid % 3;    // valid for tid < 126
    const int vv = ch / 6, cc = ch % 6;
    for (int tile = 0; tile < 8; ++tile) {
        const int n0 = (blk * 8 + tile) * 16;
        for (int idx = tid; idx < 16 * SAMP_FLOATS; idx += 256)
            lds[idx] = x[(size_t)n0 * SAMP_FLOATS + idx];
        __syncthreads();
        if (tid < 126) {
            for (int s = 0; s < 16; ++s)
                for (int t = p; t < NT; t += 3) {
                    float val = lds[s * SAMP_FLOATS + (cc * NT + t) * NV + vv];
                    sum += val; sq += val * val;
                }
        }
        __syncthreads();
    }
    if (tid < 126) { lds[tid] = sum; lds[126 + tid] = sq; }
    __syncthreads();
    if (tid < 42) {
        float s0 = lds[tid*3] + lds[tid*3+1] + lds[tid*3+2];
        float q0 = lds[126+tid*3] + lds[126+tid*3+1] + lds[126+tid*3+2];
        part[blk * 84 + tid]      = s0;
        part[blk * 84 + 42 + tid] = q0;
    }
}

// ---------------------------------------------------------------------------
// Kernel 2: finalize stats, folded biases, normalized adjacencies, misc.
// ---------------------------------------------------------------------------
__global__ void k_prep1(const float* __restrict__ part,
                        const float* __restrict__ gamma, const float* __restrict__ beta,
                        const float* __restrict__ W1, const float* __restrict__ a1,
                        const float* __restrict__ B1, const float* __restrict__ W2,
                        const float* __restrict__ a2, const float* __restrict__ B2,
                        const float* __restrict__ b_ih, const float* __restrict__ b_hh,
                        float* __restrict__ ws)
{
    const int tid = threadIdx.x;
    if (tid < 42) {                              // deterministic fixed-order reduce
        float s = 0.f, q = 0.f;
        for (int b = 0; b < 256; ++b) { s += part[b*84 + tid]; q += part[b*84 + 42 + tid]; }
        float mean = s / (float)BN_CNT;
        float var  = q / (float)BN_CNT - mean * mean;
        float inv  = rsqrtf(var + 1e-5f);
        float sc   = gamma[tid] * inv;
        ws[OFF_SCALE + tid] = sc;
        ws[OFF_SHIFT + tid] = beta[tid] - mean * sc;
    }
    __syncthreads();
    // folded bias[v][col]
    for (int idx = tid; idx < 7 * 128; idx += 256) {
        const int v = idx / 128, col = idx % 128;
        float b = 0.f;
        if (col < 54) {
            const int g = col / 9, e = col % 9, H = headOf(g);
            for (int c = 0; c < 6; ++c) {
                float cs = 0.f;
                for (int t = 0; t < NT; ++t) cs += W1[(H * 180 + c * 30 + t) * 9 + e];
                b += ws[OFF_SHIFT + v * 6 + c] * cs;
            }
        } else if (col < 114) {
            const int c = (col - 54) / 10, f = (col - 54) % 10;
            float cs = 0.f;
            for (int t = 0; t < NT; ++t) cs += W2[t * 10 + f];
            b = ws[OFF_SHIFT + v * 6 + c] * cs;
        }
        ws[OFF_BIAS + idx] = b;
    }
    // normalized adjacencies (6 GAT1 heads + GAT2 head 0)
    if (tid < 7) {
        const float* src = (tid < 6) ? (B1 + headOf(tid) * 49) : B2;
        float adj[49]; float mn = 1e30f, mx = -1e30f;
        for (int i = 0; i < 7; ++i)
            for (int j = 0; j < 7; ++j) {
                float a = src[i * 7 + j] + ((i == j) ? 1.f : 0.f);
                adj[i * 7 + j] = a; mn = fminf(mn, a); mx = fmaxf(mx, a);
            }
        float inv = 1.f / (mx - mn);
        float r[7];
        for (int i = 0; i < 7; ++i) {
            float rs = 0.f;
            for (int j = 0; j < 7; ++j) { adj[i*7+j] = (adj[i*7+j] - mn) * inv; rs += adj[i*7+j]; }
            r[i] = rsqrtf(rs);
        }
        float* dst = (tid < 6) ? (ws + OFF_ADJ1 + tid * 49) : (ws + OFF_ADJ2);
        for (int i = 0; i < 7; ++i)
            for (int j = 0; j < 7; ++j) dst[i * 7 + j] = adj[i * 7 + j] * r[i] * r[j];
    }
    // attention vectors
    for (int idx = tid; idx < 108; idx += 256) {
        const int g = idx / 18, k = idx % 18;
        ws[OFF_A1 + idx] = a1[headOf(g) * 18 + k];
    }
    if (tid < 20) ws[OFF_A2 + tid] = a2[tid];
    for (int idx = tid; idx < 96; idx += 256)
        ws[OFF_GB + idx] = (idx < 84) ? (b_ih[idx] + b_hh[idx]) : 0.f;
}

// ---------------------------------------------------------------------------
// Kernel 3: build bf16 B matrices, pre-swizzled into WMMA B-fragment layout.
// blocks 0..335: (v,kk,nn) fragments of the fused GAT weights (192x128 per v)
// blocks 336..353: W_ih^T fragments for the batched LSTM gate GEMM (96x96)
// ---------------------------------------------------------------------------
__global__ void k_prep2(const float* __restrict__ W1, const float* __restrict__ W2,
                        const float* __restrict__ W_ih, const float* __restrict__ ws,
                        __bf16* __restrict__ wsB, __bf16* __restrict__ wsWih)
{
    const int b = blockIdx.x, tid = threadIdx.x;
    if (b < 336) {
        const int v = b / 48, kk = (b / 8) % 6, nn = b % 8;
        for (int j = tid; j < 512; j += 256) {
            const int L = j >> 4, i = j & 15;
            const int k = kk * 32 + kmapB(i, L);
            const int col = nn * 16 + (L & 15);
            float val = 0.f;
            if (k < 180 && col < 114) {
                const int c = k / 30, t = k % 30;
                const float sc = ws[OFF_SCALE + v * 6 + c];
                if (col < 54) {
                    const int g = col / 9, e = col % 9;
                    val = sc * W1[(headOf(g) * 180 + k) * 9 + e];
                } else {
                    const int c2 = (col - 54) / 10, f = (col - 54) % 10;
                    val = (c == c2) ? sc * W2[t * 10 + f] : 0.f;
                }
            }
            wsB[(size_t)b * 512 + j] = (__bf16)val;
        }
    } else {
        const int idx = b - 336, kk = idx / 6, nn = idx % 6;
        for (int j = tid; j < 512; j += 256) {
            const int L = j >> 4, i = j & 15;
            const int k = kk * 32 + kmapB(i, L);
            const int col = nn * 16 + (L & 15);
            float val = (k < 84 && col < 84) ? W_ih[col * 84 + k] : 0.f;
            wsWih[(size_t)idx * 512 + j] = (__bf16)val;
        }
    }
}

// ---------------------------------------------------------------------------
// Kernel 4: fused main kernel.  256 threads (8 waves) per 16 samples.
// ---------------------------------------------------------------------------
__global__ void __launch_bounds__(256)
k_main(const float* __restrict__ x, const float* __restrict__ ws,
       const __bf16* __restrict__ wsB, const __bf16* __restrict__ wsWih,
       const float* __restrict__ W_out, const float* __restrict__ b_out,
       float* __restrict__ out)
{
    extern __shared__ float lds[];
    float* s_x    = lds;            // 16*1260 = 20160
    float* s_wh   = s_x + 20160;    // 16*7*128 = 14336
    float* s_xs0  = s_wh + 14336;   // 16*96 = 1536
    float* s_gate = s_xs0 + 1536;   // 16*96 = 1536
    float* s_h1   = s_gate + 1536;  // 16*21 = 336
    const int tid = threadIdx.x;
    const int wave = tid >> 5, lane = tid & 31;
    const int n0 = blockIdx.x * 16;

    // ---- Phase A: stage 16 samples, zero xs0 pad -------------------------
    for (int idx = tid; idx < 16 * SAMP_FLOATS; idx += 256)
        s_x[idx] = x[(size_t)n0 * SAMP_FLOATS + idx];
    for (int idx = tid; idx < 16 * 96; idx += 256) s_xs0[idx] = 0.f;
    __syncthreads();

    // ---- Phase B: per-node GEMM, wave v -> Wh[:, v, :128] ----------------
    if (wave < 7) {
        const int v = wave;
        const int m = lane & 15;
        v8f acc[8] = {};
        for (int kk = 0; kk < 6; ++kk) {
            v16bf a;
#pragma unroll
            for (int i = 0; i < 16; ++i) {
                const int k = kk * 32 + kmapA(i, lane);
                float val = (k < 180) ? s_x[m * SAMP_FLOATS + k * NV + v] : 0.f;
                a[i] = (__bf16)val;
            }
            const __bf16* fb = wsB + (size_t)((v * 6 + kk) * 8) * 512 + lane * 16;
#pragma unroll
            for (int nn = 0; nn < 8; ++nn) {
                v16bf bf = *(const v16bf*)(fb + nn * 512);
                acc[nn] = __builtin_amdgcn_wmma_f32_16x16x32_bf16(
                    false, a, false, bf, (short)0, acc[nn], false, false);
            }
        }
        const int mrow = (lane & 16) ? 8 : 0;
#pragma unroll
        for (int nn = 0; nn < 8; ++nn) {
            const int col = nn * 16 + m;
            const float bias = ws[OFF_BIAS + v * 128 + col];
#pragma unroll
            for (int r = 0; r < 8; ++r)
                s_wh[((mrow + r) * 7 + v) * 128 + col] = acc[nn][r] + bias;
        }
    }
    __syncthreads();

    // ---- Phase C: attention epilogue (2 samples/wave, 16 lanes each) -----
    {
        const int s = wave * 2 + ((lane >> 4) & 1);
        const int l = lane & 15;
        const int vv = (l < 7) ? l : 0;
        const float* wh_s = s_wh + s * 7 * 128;

        // GAT1: 6 surviving heads, one output column each
#pragma unroll
        for (int g = 0; g < 6; ++g) {
            const int cbase = g * 9;
            const int ecol  = cbase + 3 * (g % 3);
            float si = 0.f, sj = 0.f;
            if (l < 7) {
#pragma unroll
                for (int e = 0; e < 9; ++e) {
                    float w = wh_s[vv * 128 + cbase + e];
                    si += w * ws[OFF_A1 + g * 18 + e];
                    sj += w * ws[OFF_A1 + g * 18 + 9 + e];
                }
            }
            float att[7]; float mx = -1e30f;
#pragma unroll
            for (int j = 0; j < 7; ++j) {
                float t = leakyf(si + __shfl(sj, j, 16));
                att[j] = t; mx = fmaxf(mx, t);
            }
            float ssum = 0.f;
#pragma unroll
            for (int j = 0; j < 7; ++j) { att[j] = expf(att[j] - mx); ssum += att[j]; }
            const float rs = 1.f / ssum;
            float mval = 0.f;                   // row-i of softmax(att) dot Wh[:,ecol]
#pragma unroll
            for (int j = 0; j < 7; ++j) mval += att[j] * rs * wh_s[j * 128 + ecol];
            float o = 0.f;                      // adj_norm mix over rows
#pragma unroll
            for (int u = 0; u < 7; ++u)
                o += ws[OFF_ADJ1 + g * 49 + vv * 7 + u] * __shfl(mval, u, 16);
            o = eluf(o);
            if (l < 7) s_xs0[s * 96 + g * 7 + l] = o;
        }

        // GAT2 head 0, feature f=0 (softmax over channel dim)
        float si2[6], sj2[6];
#pragma unroll
        for (int c = 0; c < 6; ++c) {
            float a_ = 0.f, b_ = 0.f;
#pragma unroll
            for (int f = 0; f < 10; ++f) {
                float w = wh_s[vv * 128 + 54 + c * 10 + f];
                a_ += w * ws[OFF_A2 + f];
                b_ += w * ws[OFF_A2 + 10 + f];
            }
            si2[c] = a_; sj2[c] = b_;
        }
        float h2[6] = {0, 0, 0, 0, 0, 0};
        for (int j = 0; j < 7; ++j) {
            float e2[6]; float mx2 = -1e30f;
#pragma unroll
            for (int c = 0; c < 6; ++c) {
                float t = leakyf(si2[c] + __shfl(sj2[c], j, 16));
                e2[c] = t; mx2 = fmaxf(mx2, t);
            }
            float ss = 0.f;
#pragma unroll
            for (int c = 0; c < 6; ++c) { e2[c] = expf(e2[c] - mx2); ss += e2[c]; }
            const float inv = 1.f / ss;
#pragma unroll
            for (int c = 0; c < 6; ++c)
                h2[c] += (e2[c] * inv) * wh_s[j * 128 + 54 + c * 10];
        }
#pragma unroll
        for (int c = 0; c < 6; ++c) {
            float o = 0.f;
#pragma unroll
            for (int u = 0; u < 7; ++u)
                o += __shfl(h2[c], u, 16) * ws[OFF_ADJ2 + u * 7 + vv];
            o = eluf(o);
            if (l < 7) s_xs0[s * 96 + 42 + c * 7 + l] = o;
        }
    }
    __syncthreads();

    // ---- Phase D: batched LSTM gate GEMM (16 x 96 x 96) ------------------
    if (wave < 6) {
        const int nn = wave;
        const int m = lane & 15;
        v8f g = {};
        for (int kk = 0; kk < 3; ++kk) {
            v16bf a;
#pragma unroll
            for (int i = 0; i < 16; ++i) {
                const int k = kk * 32 + kmapA(i, lane);
                a[i] = (__bf16)s_xs0[m * 96 + k];
            }
            v16bf bf = *(const v16bf*)(wsWih + (size_t)(kk * 6 + nn) * 512 + lane * 16);
            g = __builtin_amdgcn_wmma_f32_16x16x32_bf16(
                false, a, false, bf, (short)0, g, false, false);
        }
        const int col = nn * 16 + m;
        const float gb = ws[OFF_GB + col];
        const int mrow = (lane & 16) ? 8 : 0;
#pragma unroll
        for (int r = 0; r < 8; ++r)
            s_gate[(mrow + r) * 96 + col] = g[r] + gb;
    }
    __syncthreads();

    // ---- Phase E: gate nonlinearities + output head ----------------------
    for (int idx = tid; idx < 16 * 21; idx += 256) {
        const int s = idx / 21, q = idx % 21;
        float gi = s_gate[s * 96 + q];
        float gg = s_gate[s * 96 + 42 + q];
        float go = s_gate[s * 96 + 63 + q];
        float c1 = sigmf(gi) * tanhf(gg);
        s_h1[idx] = sigmf(go) * tanhf(c1);
    }
    __syncthreads();
    for (int idx = tid; idx < 16 * 7; idx += 256) {
        const int s = idx / 7, o = idx % 7;
        float acc = b_out[o];
#pragma unroll
        for (int q = 0; q < 21; ++q) acc += s_h1[s * 21 + q] * W_out[o * 21 + q];
        out[(size_t)(n0 + s) * 7 + o] = acc;
    }
}

// ---------------------------------------------------------------------------
extern "C" void kernel_launch(void* const* d_in, const int* in_sizes, int n_in,
                              void* d_out, int out_size, void* d_ws, size_t ws_size,
                              hipStream_t stream)
{
    (void)in_sizes; (void)n_in; (void)out_size; (void)ws_size;
    const float* x     = (const float*)d_in[0];
    const float* gamma = (const float*)d_in[1];
    const float* beta  = (const float*)d_in[2];
    const float* W1    = (const float*)d_in[3];
    const float* a1    = (const float*)d_in[4];
    const float* B1    = (const float*)d_in[5];
    const float* W2    = (const float*)d_in[6];
    const float* a2    = (const float*)d_in[7];
    const float* B2    = (const float*)d_in[8];
    const float* W_ih  = (const float*)d_in[9];
    /* d_in[10] = W_hh, unused (h0 == 0 -> only step 0 matters) */
    const float* b_ih  = (const float*)d_in[11];
    const float* b_hh  = (const float*)d_in[12];
    const float* W_out = (const float*)d_in[13];
    const float* b_out = (const float*)d_in[14];

    float*  ws    = (float*)d_ws;
    __bf16* wsB   = (__bf16*)((char*)d_ws + OFFB_WSB);
    __bf16* wsWih = (__bf16*)((char*)d_ws + OFFB_WIH);
    float*  part  = (float*)((char*)d_ws + OFFB_PART);

    k_stats<<<256, 256, 16 * SAMP_FLOATS * sizeof(float), stream>>>(x, part);
    k_prep1<<<1, 256, 0, stream>>>(part, gamma, beta, W1, a1, B1, W2, a2, B2,
                                   b_ih, b_hh, ws);
    k_prep2<<<354, 256, 0, stream>>>(W1, W2, W_ih, ws, wsB, wsWih);

    const size_t lds_bytes = (20160 + 14336 + 1536 + 1536 + 336) * sizeof(float);
    k_main<<<NSAMP / 16, 256, lds_bytes, stream>>>(x, ws, wsB, wsWih, W_out, b_out,
                                                   (float*)d_out);
}
